// GraphNeuralNetwork_63170378990110
// MI455X (gfx1250) — matrix operand order repaired and checked by reading, not compiled, not run on hardware
//
#include <hip/hip_runtime.h>

#define NNODE 20000
#define NEDGE 200000
#define HID   128

typedef __attribute__((ext_vector_type(16))) _Float16 v16h;
typedef __attribute__((ext_vector_type(8)))  _Float16 v8h;
typedef __attribute__((ext_vector_type(8)))  float    v8f;

// ---------------------------------------------------------------------------
// Weight prep: convert [128 x Kreal] f32 (torch Linear [out,in]) into
// fragment-swizzled f16 B-matrix buffers for v_wmma_f32_16x16x32_f16.
// Element t of dst:  j = t&15 (half within lane), lane = (t>>4)&31,
// frag = t>>9 = kk*8+nn.  K = kk*32 + (lane>=16 ? 16 : 0) + j, N = nn*16 + lane%16.
// ---------------------------------------------------------------------------
__global__ void prep_w_kernel(const float* __restrict__ src, _Float16* __restrict__ dst,
                              int kreal, int kpad) {
  int t = blockIdx.x * 256 + threadIdx.x;
  if (t >= kpad * HID) return;
  int j    = t & 15;
  int lane = (t >> 4) & 31;
  int frag = t >> 9;
  int nn = frag & 7;
  int kk = frag >> 3;
  int k = kk * 32 + ((lane >> 4) << 4) + j;
  int n = nn * 16 + (lane & 15);
  float v = (k < kreal) ? src[(size_t)n * kreal + k] : 0.f;
  dst[t] = (_Float16)v;
}

// ---------------------------------------------------------------------------
// WMMA helpers: one wave computes a 32-row strip (two 16-row sub-strips) so
// every B fragment feeds two WMMAs; B is prefetched one nn ahead into a
// separate register set so the MMA pipe is not stalled on loadcnt 0.
// ---------------------------------------------------------------------------
template <int KSTEPS>
__device__ __forceinline__ void wmma_gemm2(const _Float16* __restrict__ Ar0,
                                           const _Float16* __restrict__ Ar1, int kcol,
                                           const v16h* __restrict__ W, int lane, int hi,
                                           v8f acc0[8], v8f acc1[8]) {
#pragma unroll
  for (int kk = 0; kk < KSTEPS; ++kk) {
    union { v16h v; v8h h[2]; } a0, a1;
    const _Float16* p0 = Ar0 + kcol + kk * 32 + hi * 16;
    const _Float16* p1 = Ar1 + kcol + kk * 32 + hi * 16;
    a0.h[0] = *(const v8h*)(p0);
    a0.h[1] = *(const v8h*)(p0 + 8);
    a1.h[0] = *(const v8h*)(p1);
    a1.h[1] = *(const v8h*)(p1 + 8);
    const v16h* wp = W + (size_t)(kk * 8) * 32 + lane;
    v16h b = wp[0];
#pragma unroll
    for (int nn = 0; nn < 8; ++nn) {
      v16h bnext = b;
      if (nn < 7) bnext = wp[(size_t)(nn + 1) * 32];   // prefetch next B frag
      acc0[nn] = __builtin_amdgcn_wmma_f32_16x16x32_f16(
          false, a0.v, false, b, (short)0, acc0[nn], false, false);
      acc1[nn] = __builtin_amdgcn_wmma_f32_16x16x32_f16(
          false, a1.v, false, b, (short)0, acc1[nn], false, false);
      b = bnext;
    }
  }
}

__device__ __forceinline__ void zero_acc(v8f acc[8]) {
  v8f z = {};
#pragma unroll
  for (int i = 0; i < 8; ++i) acc[i] = z;
}

// D element (acc[nn][r]) = tile row (rowoff + r + hi*8), col (nn*16 + lane%16).
__device__ __forceinline__ void epilog_relu_lds(const v8f acc[8], const float* __restrict__ bias,
                                                _Float16* Aw, int kpad, int dstcol,
                                                int l16, int hi, int rowoff) {
#pragma unroll
  for (int nn = 0; nn < 8; ++nn) {
    float bv = bias[nn * 16 + l16];
#pragma unroll
    for (int r = 0; r < 8; ++r) {
      float v = acc[nn][r] + bv;
      v = v > 0.f ? v : 0.f;
      Aw[(rowoff + r + hi * 8) * kpad + dstcol + nn * 16 + l16] = (_Float16)v;
    }
  }
}

// ---------------------------------------------------------------------------
// Encoder + fuse:  mean-over-L of Conv1d(8->128,k=3,pad=1) collapses to a
// 24-term dot against per-channel prefix sums; then Linear(129->128).
// Block = 256 threads = 2 nodes x 128 output features.
// ---------------------------------------------------------------------------
__global__ __launch_bounds__(256) void encode_kernel(
    const float* __restrict__ x, const float* __restrict__ node_type,
    const float* __restrict__ emb, const float* __restrict__ cw,
    const float* __restrict__ cb, const float* __restrict__ fw,
    const float* __restrict__ fb, _Float16* __restrict__ h_nodes) {
  __shared__ float s[2][24];
  __shared__ float hb[2][HID];
  int sub = threadIdx.x >> 7;
  int o   = threadIdx.x & 127;
  int n   = blockIdx.x * 2 + sub;
  if (o < 8) {
    int i = o;
    float v0, v1, v2, v3, v4;
    if (i < 4) {
      const float* xp = x + (size_t)n * 20 + i;   // x[n][t][i], stride 4 over t
      v0 = xp[0]; v1 = xp[4]; v2 = xp[8]; v3 = xp[12]; v4 = xp[16];
    } else {
      const float* ep = emb + (i - 4);            // emb[t][i-4]
      v0 = ep[0]; v1 = ep[4]; v2 = ep[8]; v3 = ep[12]; v4 = ep[16];
    }
    float a = v0 + v1 + v2 + v3;   // k=0: t sums over x[0..3]
    float b = a + v4;              // k=1: x[0..4]
    float c = b - v0;              // k=2: x[1..4]
    s[sub][i * 3 + 0] = a * 0.2f;
    s[sub][i * 3 + 1] = b * 0.2f;
    s[sub][i * 3 + 2] = c * 0.2f;
  }
  __syncthreads();
  float h = cb[o];
  const float* w = cw + o * 24;    // conv_w[o][i][k] contiguous 24 floats
#pragma unroll
  for (int q = 0; q < 24; ++q) h += w[q] * s[sub][q];
  hb[sub][o] = h;
  __syncthreads();
  const float* f = fw + (size_t)o * 129;
  float h2 = fb[o] + f[128] * node_type[n];
#pragma unroll 8
  for (int q = 0; q < 128; ++q) h2 += f[q] * hb[sub][q];
  h_nodes[(size_t)n * HID + o] = (_Float16)h2;
}

// ---------------------------------------------------------------------------
// gnn1: edge MLP (K=260 pad 288 -> 128 -> 128) fused with node MLP
// (K=256 -> 128 -> 128).  Tile = 64 edges (2 waves x 32-row strips).
// LDS A-tile column reuse:
//   stage [src|dst|ea_pad]  cols 0..287
//   e-hidden -> cols 160..287 ; ea -> cols 128..255 (also to global f16)
//   n-hidden -> cols 0..127   ; h_out -> global f16
// All LDS traffic is wave-private -> no barriers.
// ---------------------------------------------------------------------------
#define KPAD1 296
__global__ __launch_bounds__(64) void gnn1_kernel(
    const _Float16* __restrict__ h_nodes, const int* __restrict__ ei,
    const float* __restrict__ eattr,
    const _Float16* __restrict__ w_e1, const float* __restrict__ b_e1,
    const _Float16* __restrict__ w_e2, const float* __restrict__ b_e2,
    const _Float16* __restrict__ w_n1, const float* __restrict__ b_n1,
    const _Float16* __restrict__ w_n2, const float* __restrict__ b_n2,
    _Float16* __restrict__ ea_out, _Float16* __restrict__ h_out) {
  __shared__ __align__(16) _Float16 A[64 * KPAD1];
  int tid = threadIdx.x, wave = tid >> 5, lane = tid & 31;
  int l16 = lane & 15, hi = lane >> 4;
  _Float16* Aw = A + wave * 32 * KPAD1;
  size_t e0 = (size_t)blockIdx.x * 64 + wave * 32;

  for (int r = 0; r < 32; ++r) {
    size_t e = e0 + r;
    int src = ei[e], dst = ei[NEDGE + e];
    ((uint2*)(Aw + r * KPAD1))[lane]       = ((const uint2*)(h_nodes + (size_t)src * HID))[lane];
    ((uint2*)(Aw + r * KPAD1 + 128))[lane] = ((const uint2*)(h_nodes + (size_t)dst * HID))[lane];
    float fv = 0.f;
    if (lane < 4) fv = eattr[e * 4 + lane];
    Aw[r * KPAD1 + 256 + lane] = (_Float16)fv;   // ea padded 4 -> 32
  }

  const _Float16* Ar0 = Aw + l16 * KPAD1;
  const _Float16* Ar1 = Ar0 + 16 * KPAD1;
  v8f acc0[8], acc1[8];

  // edge GEMM1: K=288
  zero_acc(acc0); zero_acc(acc1);
  wmma_gemm2<9>(Ar0, Ar1, 0, (const v16h*)w_e1, lane, hi, acc0, acc1);
  epilog_relu_lds(acc0, b_e1, Aw, KPAD1, 160, l16, hi, 0);
  epilog_relu_lds(acc1, b_e1, Aw, KPAD1, 160, l16, hi, 16);

  // edge GEMM2: K=128 (hidden at col 160) -> ea
  zero_acc(acc0); zero_acc(acc1);
  wmma_gemm2<4>(Ar0, Ar1, 160, (const v16h*)w_e2, lane, hi, acc0, acc1);
#pragma unroll
  for (int s = 0; s < 2; ++s) {
    const v8f* ac = s ? acc1 : acc0;
    int rowoff = s * 16;
#pragma unroll
    for (int nn = 0; nn < 8; ++nn) {
      float bv = b_e2[nn * 16 + l16];
#pragma unroll
      for (int r = 0; r < 8; ++r) {
        _Float16 hv = (_Float16)(ac[nn][r] + bv);
        Aw[(rowoff + r + hi * 8) * KPAD1 + 128 + nn * 16 + l16] = hv;
        ea_out[(e0 + rowoff + r + hi * 8) * HID + nn * 16 + l16] = hv;
      }
    }
  }

  // node GEMM1: K=256 ([src|ea] at cols 0..255)
  zero_acc(acc0); zero_acc(acc1);
  wmma_gemm2<8>(Ar0, Ar1, 0, (const v16h*)w_n1, lane, hi, acc0, acc1);
  epilog_relu_lds(acc0, b_n1, Aw, KPAD1, 0, l16, hi, 0);
  epilog_relu_lds(acc1, b_n1, Aw, KPAD1, 0, l16, hi, 16);

  // node GEMM2: K=128 -> h_out
  zero_acc(acc0); zero_acc(acc1);
  wmma_gemm2<4>(Ar0, Ar1, 0, (const v16h*)w_n2, lane, hi, acc0, acc1);
#pragma unroll
  for (int s = 0; s < 2; ++s) {
    const v8f* ac = s ? acc1 : acc0;
    int rowoff = s * 16;
#pragma unroll
    for (int nn = 0; nn < 8; ++nn) {
      float bv = b_n2[nn * 16 + l16];
#pragma unroll
      for (int r = 0; r < 8; ++r)
        h_out[(e0 + rowoff + r + hi * 8) * HID + nn * 16 + l16] = (_Float16)(ac[nn][r] + bv);
    }
  }
}

// ---------------------------------------------------------------------------
// gnn2: edge MLP (K=384 -> 128 -> 128) fused with final z = Linear(128->128),
// plus per-feature sum/sumsq accumulation for BatchNorm.
// (The reference's second node MLP output is dead code -> skipped.)
// ---------------------------------------------------------------------------
#define KPAD2 392
__global__ __launch_bounds__(64) void gnn2_kernel(
    const _Float16* __restrict__ h_edges, const int* __restrict__ ei,
    const _Float16* __restrict__ ea1,
    const _Float16* __restrict__ w_e1, const float* __restrict__ b_e1,
    const _Float16* __restrict__ w_e2, const float* __restrict__ b_e2,
    const _Float16* __restrict__ w_z,  const float* __restrict__ b_z,
    float* __restrict__ z_out, float* __restrict__ gsum, float* __restrict__ gsq) {
  __shared__ __align__(16) _Float16 A[64 * KPAD2];
  __shared__ float lsum[HID], lsq[HID];
  int tid = threadIdx.x, wave = tid >> 5, lane = tid & 31;
  int l16 = lane & 15, hi = lane >> 4;
  if (tid < 64) {
    lsum[tid] = 0.f; lsq[tid] = 0.f;
    lsum[tid + 64] = 0.f; lsq[tid + 64] = 0.f;
  }
  __syncthreads();

  _Float16* Aw = A + wave * 32 * KPAD2;
  size_t e0 = (size_t)blockIdx.x * 64 + wave * 32;

  for (int r = 0; r < 32; ++r) {
    size_t e = e0 + r;
    int src = ei[e], dst = ei[NEDGE + e];
    ((uint2*)(Aw + r * KPAD2))[lane]       = ((const uint2*)(h_edges + (size_t)src * HID))[lane];
    ((uint2*)(Aw + r * KPAD2 + 128))[lane] = ((const uint2*)(h_edges + (size_t)dst * HID))[lane];
    ((uint2*)(Aw + r * KPAD2 + 256))[lane] = ((const uint2*)(ea1 + e * HID))[lane];
  }

  const _Float16* Ar0 = Aw + l16 * KPAD2;
  const _Float16* Ar1 = Ar0 + 16 * KPAD2;
  v8f acc0[8], acc1[8];

  // edge GEMM1: K=384
  zero_acc(acc0); zero_acc(acc1);
  wmma_gemm2<12>(Ar0, Ar1, 0, (const v16h*)w_e1, lane, hi, acc0, acc1);
  epilog_relu_lds(acc0, b_e1, Aw, KPAD2, 0, l16, hi, 0);
  epilog_relu_lds(acc1, b_e1, Aw, KPAD2, 0, l16, hi, 16);

  // edge GEMM2: K=128 -> ea2 (LDS cols 128..255 only; never needed globally)
  zero_acc(acc0); zero_acc(acc1);
  wmma_gemm2<4>(Ar0, Ar1, 0, (const v16h*)w_e2, lane, hi, acc0, acc1);
#pragma unroll
  for (int s = 0; s < 2; ++s) {
    const v8f* ac = s ? acc1 : acc0;
    int rowoff = s * 16;
#pragma unroll
    for (int nn = 0; nn < 8; ++nn) {
      float bv = b_e2[nn * 16 + l16];
#pragma unroll
      for (int r = 0; r < 8; ++r)
        Aw[(rowoff + r + hi * 8) * KPAD2 + 128 + nn * 16 + l16] = (_Float16)(ac[nn][r] + bv);
    }
  }

  // z GEMM: K=128 from col 128 -> z (f32) + BN stats
  zero_acc(acc0); zero_acc(acc1);
  wmma_gemm2<4>(Ar0, Ar1, 128, (const v16h*)w_z, lane, hi, acc0, acc1);
#pragma unroll
  for (int nn = 0; nn < 8; ++nn) {
    float bv = b_z[nn * 16 + l16];
    float ps = 0.f, pq = 0.f;
#pragma unroll
    for (int s = 0; s < 2; ++s) {
      const v8f* ac = s ? acc1 : acc0;
      int rowoff = s * 16;
#pragma unroll
      for (int r = 0; r < 8; ++r) {
        float v = ac[nn][r] + bv;
        z_out[(e0 + rowoff + r + hi * 8) * HID + nn * 16 + l16] = v;
        ps += v;
        pq += v * v;
      }
    }
    ps += __shfl_xor(ps, 16, 32);
    pq += __shfl_xor(pq, 16, 32);
    if (hi == 0) {
      atomicAdd(&lsum[nn * 16 + l16], ps);
      atomicAdd(&lsq[nn * 16 + l16], pq);
    }
  }
  __syncthreads();
  if (tid < 64) {
    atomicAdd(&gsum[tid], lsum[tid]);
    atomicAdd(&gsq[tid],  lsq[tid]);
    atomicAdd(&gsum[tid + 64], lsum[tid + 64]);
    atomicAdd(&gsq[tid + 64],  lsq[tid + 64]);
  }
}

// ---------------------------------------------------------------------------
// BN finalize: scale/shift per feature (biased variance, eps=1e-5).
// ---------------------------------------------------------------------------
__global__ void bn_finalize_kernel(const float* __restrict__ gsum, const float* __restrict__ gsq,
                                   const float* __restrict__ bn_g, const float* __restrict__ bn_b,
                                   float* __restrict__ scale, float* __restrict__ shift) {
  int i = threadIdx.x;
  float inv = 1.f / (float)NEDGE;
  float mu  = gsum[i] * inv;
  float var = gsq[i] * inv - mu * mu;
  float sc  = bn_g[i] * rsqrtf(var + 1e-5f);
  scale[i] = sc;
  shift[i] = bn_b[i] - mu * sc;
}

// ---------------------------------------------------------------------------
// Head: out = relu(z*scale+shift) @ fm_w2^T + fm_b2.  One wave per edge,
// lane owns 4 features, wave shuffle reduce for the 3 outputs.
// ---------------------------------------------------------------------------
__global__ __launch_bounds__(256) void head_kernel(
    const float* __restrict__ z, const float* __restrict__ scale,
    const float* __restrict__ shift, const float* __restrict__ w2,
    const float* __restrict__ b2, float* __restrict__ out) {
  int tid = threadIdx.x, wave = tid >> 5, lane = tid & 31;
  float wa[4], wb[4], wc[4], sc[4], sh[4];
#pragma unroll
  for (int c = 0; c < 4; ++c) {
    int o = lane * 4 + c;
    wa[c] = w2[o];
    wb[c] = w2[128 + o];
    wc[c] = w2[256 + o];
    sc[c] = scale[o];
    sh[c] = shift[o];
  }
  float bb0 = b2[0], bb1 = b2[1], bb2 = b2[2];
  size_t stride = (size_t)gridDim.x * 8;
  for (size_t e = (size_t)blockIdx.x * 8 + wave; e < NEDGE; e += stride) {
    const float4 zz = *(const float4*)(z + e * HID + lane * 4);
    float vv[4] = {zz.x, zz.y, zz.z, zz.w};
    float a0 = 0.f, a1 = 0.f, a2 = 0.f;
#pragma unroll
    for (int c = 0; c < 4; ++c) {
      float v = vv[c] * sc[c] + sh[c];
      v = v > 0.f ? v : 0.f;
      a0 += v * wa[c];
      a1 += v * wb[c];
      a2 += v * wc[c];
    }
#pragma unroll
    for (int off = 16; off > 0; off >>= 1) {
      a0 += __shfl_xor(a0, off, 32);
      a1 += __shfl_xor(a1, off, 32);
      a2 += __shfl_xor(a2, off, 32);
    }
    if (lane == 0) {
      out[e * 3 + 0] = a0 + bb0;
      out[e * 3 + 1] = a1 + bb1;
      out[e * 3 + 2] = a2 + bb2;
    }
  }
}

// ---------------------------------------------------------------------------
extern "C" void kernel_launch(void* const* d_in, const int* in_sizes, int n_in,
                              void* d_out, int out_size, void* d_ws, size_t ws_size,
                              hipStream_t stream) {
  const float* x      = (const float*)d_in[0];
  const int*   ei     = (const int*)d_in[1];
  const float* eattr  = (const float*)d_in[2];
  /* d_in[3] = batch (unused) */
  const float* ntype  = (const float*)d_in[4];
  const float* emb    = (const float*)d_in[5];
  const float* conv_w = (const float*)d_in[6];
  const float* conv_b = (const float*)d_in[7];
  const float* fuse_w = (const float*)d_in[8];
  const float* fuse_b = (const float*)d_in[9];
  const float* g1e_w1 = (const float*)d_in[10]; const float* g1e_b1 = (const float*)d_in[11];
  const float* g1e_w2 = (const float*)d_in[12]; const float* g1e_b2 = (const float*)d_in[13];
  const float* g1n_w1 = (const float*)d_in[14]; const float* g1n_b1 = (const float*)d_in[15];
  const float* g1n_w2 = (const float*)d_in[16]; const float* g1n_b2 = (const float*)d_in[17];
  const float* g2e_w1 = (const float*)d_in[18]; const float* g2e_b1 = (const float*)d_in[19];
  const float* g2e_w2 = (const float*)d_in[20]; const float* g2e_b2 = (const float*)d_in[21];
  /* d_in[22..25] = g2n_* : dead in the reference, skipped */
  const float* fm_w1  = (const float*)d_in[26]; const float* fm_b1  = (const float*)d_in[27];
  const float* bn_g   = (const float*)d_in[28]; const float* bn_b   = (const float*)d_in[29];
  const float* fm_w2  = (const float*)d_in[30]; const float* fm_b2  = (const float*)d_in[31];
  float* out = (float*)d_out;

  char* ws = (char*)d_ws;
  size_t off = 0;
  auto take = [&](size_t bytes) -> void* {
    void* p = ws + off;
    off += (bytes + 255) & ~(size_t)255;
    return p;
  };
  _Float16* Wg1e1   = (_Float16*)take((size_t)288 * 128 * 2);
  _Float16* Wg1e2   = (_Float16*)take((size_t)128 * 128 * 2);
  _Float16* Wg1n1   = (_Float16*)take((size_t)256 * 128 * 2);
  _Float16* Wg1n2   = (_Float16*)take((size_t)128 * 128 * 2);
  _Float16* Wg2e1   = (_Float16*)take((size_t)384 * 128 * 2);
  _Float16* Wg2e2   = (_Float16*)take((size_t)128 * 128 * 2);
  _Float16* Wfm1    = (_Float16*)take((size_t)128 * 128 * 2);
  _Float16* h_nodes = (_Float16*)take((size_t)NNODE * HID * 2);
  _Float16* ea1     = (_Float16*)take((size_t)NEDGE * HID * 2);
  _Float16* h_edges = (_Float16*)take((size_t)NEDGE * HID * 2);
  float*    zbuf    = (float*)take((size_t)NEDGE * HID * 4);
  float*    gsum    = (float*)take(128 * 4);
  float*    gsq     = (float*)take(128 * 4);
  float*    scale   = (float*)take(128 * 4);
  float*    shift   = (float*)take(128 * 4);

  hipMemsetAsync(gsum, 0, 128 * 4, stream);
  hipMemsetAsync(gsq,  0, 128 * 4, stream);

  auto prep = [&](const float* s, _Float16* d, int kreal, int kpad) {
    int total = kpad * HID;
    prep_w_kernel<<<(total + 255) / 256, 256, 0, stream>>>(s, d, kreal, kpad);
  };
  prep(g1e_w1, Wg1e1, 260, 288);
  prep(g1e_w2, Wg1e2, 128, 128);
  prep(g1n_w1, Wg1n1, 256, 256);
  prep(g1n_w2, Wg1n2, 128, 128);
  prep(g2e_w1, Wg2e1, 384, 384);
  prep(g2e_w2, Wg2e2, 128, 128);
  prep(fm_w1,  Wfm1,  128, 128);

  encode_kernel<<<NNODE / 2, 256, 0, stream>>>(x, ntype, emb, conv_w, conv_b,
                                               fuse_w, fuse_b, h_nodes);
  gnn1_kernel<<<NEDGE / 64, 64, 0, stream>>>(h_nodes, ei, eattr,
      Wg1e1, g1e_b1, Wg1e2, g1e_b2, Wg1n1, g1n_b1, Wg1n2, g1n_b2, ea1, h_edges);
  gnn2_kernel<<<NEDGE / 64, 64, 0, stream>>>(h_edges, ei, ea1,
      Wg2e1, g2e_b1, Wg2e2, g2e_b2, Wfm1, fm_b1, zbuf, gsum, gsq);
  bn_finalize_kernel<<<1, 128, 0, stream>>>(gsum, gsq, bn_g, bn_b, scale, shift);
  head_kernel<<<3125, 256, 0, stream>>>(zbuf, scale, shift, fm_w2, fm_b2, out);

  (void)in_sizes; (void)n_in; (void)out_size; (void)ws_size;
}
